// GCN_19963007992112
// MI455X (gfx1250) — matrix operand order repaired and checked by reading, not compiled, not run on hardware
//
#include <hip/hip_runtime.h>
#include <math.h>

// ---------------- types ----------------
typedef __attribute__((ext_vector_type(16))) __bf16 v16bf;
typedef __attribute__((ext_vector_type(8)))  __bf16 v8bf;
typedef __attribute__((ext_vector_type(8)))  float  v8f;

#define N_NODES 65536
#define F_IN    128
#define EMB     256
#define TGT     128
#define N_GRAPH 256
#define NLAYERS 4

// ---------------- WMMA fragment loaders ----------------
// A fragment: 16x32 bf16 from row-major bf16 A[M][lda]; two 16B loads per lane
static __device__ inline v16bf load_a_bf(const __bf16* __restrict__ A, int lda,
                                         int rowBase, int k0, int lane) {
  int m  = lane & 15;
  int kh = (lane >> 4) << 3;                       // 0 or 8
  const __bf16* p = A + (size_t)(rowBase + m) * lda + k0 + kh;
  v16bf a;
  ((v8bf*)&a)[0] = *(const v8bf*)p;                // k = k0+kh+0..7      (VGPR0-3)
  ((v8bf*)&a)[1] = *(const v8bf*)(p + 16);         // k = k0+16+kh+0..7   (VGPR4-7)
  return a;
}

// A fragment from f32 source (A1 of k_gemm_enc: agg panel or pooled summary)
static __device__ inline v16bf load_a_f32(const float* __restrict__ A, int lda,
                                          int rowBase, int k0, int lane) {
  int m  = lane & 15;
  int kh = (lane >> 4) << 3;
  const float* p = A + (size_t)(rowBase + m) * lda + k0 + kh;
  v16bf a;
#pragma unroll
  for (int i = 0; i < 8; ++i) {
    a[i]     = (__bf16)p[i];
    a[i + 8] = (__bf16)p[i + 16];
  }
  return a;
}

// B fragment: 32x16 bf16 from row-major bf16 B[K][ldb]; lane%16 = column n, elements over k (same order as A)
static __device__ inline v16bf load_b_bf(const __bf16* __restrict__ B, int ldb,
                                         int colBase, int k0, int lane) {
  int n  = lane & 15;
  int kh = (lane >> 4) << 3;
  const __bf16* p = B + (size_t)(k0 + kh) * ldb + colBase + n;
  v16bf b;
#pragma unroll
  for (int i = 0; i < 8; ++i) {
    b[i]     = p[(size_t)i * ldb];
    b[i + 8] = p[(size_t)(i + 16) * ldb];
  }
  return b;
}

// ---------------- kernels ----------------
__global__ void k_zero(float* __restrict__ p, size_t n) {
  size_t t = (size_t)blockIdx.x * blockDim.x + threadIdx.x;
  if (t < n) p[t] = 0.0f;
}

__global__ void k_f32_to_bf16(const float* __restrict__ in, __bf16* __restrict__ out, size_t n) {
  size_t t = (size_t)blockIdx.x * blockDim.x + threadIdx.x;
  if (t < n) out[t] = (__bf16)in[t];
}

// agg[dst] += x[src] : bf16 gathers (16B), f32 atomic accumulation
__global__ void k_scatter_bf(const __bf16* __restrict__ x, const int* __restrict__ src,
                             const int* __restrict__ dst, float* __restrict__ agg,
                             int nE, int feat) {
  int cpe = feat >> 3;                             // 8-element bf16 chunks per edge
  size_t total = (size_t)nE * (size_t)cpe;
  for (size_t t = (size_t)blockIdx.x * blockDim.x + threadIdx.x; t < total;
       t += (size_t)gridDim.x * blockDim.x) {
    size_t e = t / cpe;
    int    c = (int)(t - e * cpe);
    int s = src[e], d = dst[e];
    v8bf v = *(const v8bf*)(x + (size_t)s * feat + (size_t)c * 8);
    float* o = agg + (size_t)d * feat + (size_t)c * 8;
#pragma unroll
    for (int i = 0; i < 8; ++i) atomicAdd(o + i, (float)v[i]);
  }
}

// C[M,N] = A1(f32)[M,K1]@B1[K1,N] (+ A2(bf16)[M,K2]@B2[K2,N]) + bias
__global__ void k_gemm_enc(const float* __restrict__ A1, const __bf16* __restrict__ B1, int K1,
                           const __bf16* __restrict__ A2, const __bf16* __restrict__ B2, int K2,
                           const float* __restrict__ bias, float* __restrict__ C,
                           int M, int N) {
  int lane = threadIdx.x & 31;
  int wave = threadIdx.x >> 5;
  int tilesN = N >> 4;
  int totalTiles = (M >> 4) * tilesN;
  int gw = blockIdx.x * (blockDim.x >> 5) + wave;
  if (gw >= totalTiles) return;
  int tM = gw / tilesN, tN = gw - tM * tilesN;
  int rowBase = tM << 4, colBase = tN << 4;

  v8f acc;
  float binit = bias ? bias[colBase + (lane & 15)] : 0.0f;
#pragma unroll
  for (int i = 0; i < 8; ++i) acc[i] = binit;

  for (int k0 = 0; k0 < K1; k0 += 32) {
    v16bf a = load_a_f32(A1, K1, rowBase, k0, lane);
    v16bf b = load_b_bf(B1, N, colBase, k0, lane);
    acc = __builtin_amdgcn_wmma_f32_16x16x32_bf16(false, a, false, b, (short)0, acc, false, false);
  }
  if (A2) {
    for (int k0 = 0; k0 < K2; k0 += 32) {
      v16bf a = load_a_bf(A2, K2, rowBase, k0, lane);
      v16bf b = load_b_bf(B2, N, colBase, k0, lane);
      acc = __builtin_amdgcn_wmma_f32_16x16x32_bf16(false, a, false, b, (short)0, acc, false, false);
    }
  }
  int n  = lane & 15;
  int mo = (lane >> 4) << 3;
  float* cp = C + (size_t)(rowBase + mo) * N + colBase + n;
#pragma unroll
  for (int r = 0; r < 8; ++r) cp[(size_t)r * N] = acc[r];
}

// C[M,N] = A(bf16)[M,K]@B(bf16)[K,N]  (discriminator ws = W_disc @ summary)
__global__ void k_gemm_bf(const __bf16* __restrict__ A, const __bf16* __restrict__ B, int K,
                          float* __restrict__ C, int M, int N) {
  int lane = threadIdx.x & 31;
  int wave = threadIdx.x >> 5;
  int tilesN = N >> 4;
  int totalTiles = (M >> 4) * tilesN;
  int gw = blockIdx.x * (blockDim.x >> 5) + wave;
  if (gw >= totalTiles) return;
  int tM = gw / tilesN, tN = gw - tM * tilesN;
  int rowBase = tM << 4, colBase = tN << 4;

  v8f acc = {};
  for (int k0 = 0; k0 < K; k0 += 32) {
    v16bf a = load_a_bf(A, K, rowBase, k0, lane);
    v16bf b = load_b_bf(B, N, colBase, k0, lane);
    acc = __builtin_amdgcn_wmma_f32_16x16x32_bf16(false, a, false, b, (short)0, acc, false, false);
  }
  int n  = lane & 15;
  int mo = (lane >> 4) << 3;
  float* cp = C + (size_t)(rowBase + mo) * N + colBase + n;
#pragma unroll
  for (int r = 0; r < 8; ++r) cp[(size_t)r * N] = acc[r];
}

// per-column sum & sumsq over rows; blockDim.x must equal cols
__global__ void k_bn_stats(const float* __restrict__ X, int rows, int cols,
                           float* __restrict__ sum, float* __restrict__ sumsq) {
  int c = threadIdx.x;
  float s = 0.0f, q = 0.0f;
  for (int r = blockIdx.x; r < rows; r += gridDim.x) {
    float v = X[(size_t)r * cols + c];
    s += v; q += v * v;
  }
  atomicAdd(&sum[c], s);
  atomicAdd(&sumsq[c], q);
}

// v = relu(bn(X)); optionally write bf16 H, f32 Hf, and f32 residual tot
__global__ void k_bn_relu(const float* __restrict__ X, __bf16* __restrict__ Hbf,
                          float* __restrict__ Hf, float* __restrict__ tot,
                          const float* __restrict__ sum, const float* __restrict__ sumsq,
                          const float* __restrict__ g, const float* __restrict__ b,
                          int rows, int cols, int accumulate) {
  size_t total = (size_t)rows * cols;
  float invR = 1.0f / (float)rows;
  for (size_t t = (size_t)blockIdx.x * blockDim.x + threadIdx.x; t < total;
       t += (size_t)gridDim.x * blockDim.x) {
    int c = (int)(t % cols);
    float m  = sum[c] * invR;
    float vr = sumsq[c] * invR - m * m;
    float rs = rsqrtf(vr + 1e-5f);
    float v  = (X[t] - m) * rs * g[c] + b[c];
    v = v > 0.0f ? v : 0.0f;
    if (Hbf) Hbf[t] = (__bf16)v;
    if (Hf)  Hf[t]  = v;
    if (tot) tot[t] = accumulate ? (tot[t] + v) : v;
  }
}

__global__ void k_pool_accum(const float* __restrict__ z, const int* __restrict__ batch,
                             float* __restrict__ sums, int nNodes, int cols) {
  size_t total = (size_t)nNodes * cols;
  for (size_t t = (size_t)blockIdx.x * blockDim.x + threadIdx.x; t < total;
       t += (size_t)gridDim.x * blockDim.x) {
    int nd = (int)(t / cols);
    int c  = (int)(t - (size_t)nd * cols);
    atomicAdd(&sums[(size_t)batch[nd] * cols + c], z[t]);
  }
}

__global__ void k_pool_counts(const int* __restrict__ batch, float* __restrict__ cnts, int nNodes) {
  int t = blockIdx.x * blockDim.x + threadIdx.x;
  if (t < nNodes) atomicAdd(&cnts[batch[t]], 1.0f);
}

__global__ void k_pool_div(const float* __restrict__ sums, const float* __restrict__ cnts,
                           float* __restrict__ summary, __bf16* __restrict__ summary_bf,
                           int G, int cols) {
  int t = blockIdx.x * blockDim.x + threadIdx.x;
  if (t < G * cols) {
    float v = sums[t] / fmaxf(cnts[t / cols], 1.0f);
    summary[t] = v;
    summary_bf[t] = (__bf16)v;
  }
}

// fused: S = A[M,K]@B[K,N] (bf16 in, f32 acc); reduce -log(sigmoid(+/-s)+EPS)
__global__ void k_loss_gemm(const __bf16* __restrict__ A, const __bf16* __restrict__ B,
                            int M, int N, int K, float* __restrict__ acc, int negMode) {
  int lane = threadIdx.x & 31;
  int wave = threadIdx.x >> 5;
  int tilesN = N >> 4;
  int totalTiles = (M >> 4) * tilesN;
  int gw = blockIdx.x * (blockDim.x >> 5) + wave;
  if (gw >= totalTiles) return;
  int tM = gw / tilesN, tN = gw - tM * tilesN;
  int rowBase = tM << 4, colBase = tN << 4;

  v8f c = {};
  for (int k0 = 0; k0 < K; k0 += 32) {
    v16bf a = load_a_bf(A, K, rowBase, k0, lane);
    v16bf b = load_b_bf(B, N, colBase, k0, lane);
    c = __builtin_amdgcn_wmma_f32_16x16x32_bf16(false, a, false, b, (short)0, c, false, false);
  }
  float s = 0.0f;
#pragma unroll
  for (int r = 0; r < 8; ++r) {
    float v = c[r];
    if (negMode) v = -v;                           // 1 - sigmoid(s) == sigmoid(-s)
    float sg = 1.0f / (1.0f + __expf(-v));
    s += -__logf(sg + 1e-15f);
  }
#pragma unroll
  for (int off = 16; off > 0; off >>= 1) s += __shfl_down(s, off, 32);
  if (lane == 0) atomicAdd(acc, s);
}

__global__ void k_finalize_loss(const float* __restrict__ acc, float* __restrict__ out, float scale) {
  if (threadIdx.x == 0 && blockIdx.x == 0) out[0] = acc[0] * scale;
}

// ---------------- host side ----------------
struct Bufs {
  // f32
  float *agg, *lin, *totP, *totN;
  float *sums, *summary, *wsb, *outlin;
  float *stat_sum, *stat_sq, *cnts, *lossA;
  // bf16
  __bf16 *x_bf, *xneg_bf, *h_bf, *tot_bf, *summary_bf, *wsb_bf;
  __bf16 *Wir_bf, *Wirt_bf, *Wmr_bf, *Wmrt_bf, *Wout_bf, *Wdisc_bf;
};

static void launch_zero(float* p, size_t n, hipStream_t s) {
  k_zero<<<(unsigned)((n + 255) / 256), 256, 0, s>>>(p, n);
}
static void launch_cvt(const float* in, __bf16* out, size_t n, hipStream_t s) {
  k_f32_to_bf16<<<(unsigned)((n + 255) / 256), 256, 0, s>>>(in, out, n);
}
static int gemm_blocks(int M, int N) { return ((M >> 4) * (N >> 4) + 7) / 8; }

static void run_encoder(const __bf16* x_bf, const float* x_unused,
                        const int* src, const int* dst, int nE,
                        void* const* d_in, const Bufs& B, float* tot, hipStream_t s) {
  (void)x_unused;
  const float* b_in_rel  = (const float*)d_in[7];
  const float* bn_in_g   = (const float*)d_in[9];
  const float* bn_in_b   = (const float*)d_in[10];
  const float* b_mid_rel = (const float*)d_in[12];
  const float* bn_mid_g  = (const float*)d_in[14];
  const float* bn_mid_b  = (const float*)d_in[15];

  // ---- input layer (feat = F_IN) ----
  launch_zero(B.agg, (size_t)N_NODES * F_IN, s);
  k_scatter_bf<<<8192, 256, 0, s>>>(x_bf, src, dst, B.agg, nE, F_IN);
  k_gemm_enc<<<gemm_blocks(N_NODES, EMB), 256, 0, s>>>(
      B.agg, B.Wir_bf, F_IN, x_bf, B.Wirt_bf, F_IN, b_in_rel, B.lin, N_NODES, EMB);
  launch_zero(B.stat_sum, 512, s);
  k_bn_stats<<<1024, EMB, 0, s>>>(B.lin, N_NODES, EMB, B.stat_sum, B.stat_sq);
  k_bn_relu<<<8192, 256, 0, s>>>(B.lin, B.h_bf, nullptr, tot, B.stat_sum, B.stat_sq,
                                 bn_in_g, bn_in_b, N_NODES, EMB, 0);
  // ---- mid layers (feat = EMB) ----
  for (int i = 0; i < NLAYERS; ++i) {
    launch_zero(B.agg, (size_t)N_NODES * EMB, s);
    k_scatter_bf<<<8192, 256, 0, s>>>(B.h_bf, src, dst, B.agg, nE, EMB);
    k_gemm_enc<<<gemm_blocks(N_NODES, EMB), 256, 0, s>>>(
        B.agg, B.Wmr_bf + (size_t)i * EMB * EMB, EMB,
        B.h_bf, B.Wmrt_bf + (size_t)i * EMB * EMB, EMB,
        b_mid_rel + (size_t)i * EMB, B.lin, N_NODES, EMB);
    launch_zero(B.stat_sum, 512, s);
    k_bn_stats<<<1024, EMB, 0, s>>>(B.lin, N_NODES, EMB, B.stat_sum, B.stat_sq);
    k_bn_relu<<<8192, 256, 0, s>>>(B.lin, B.h_bf, nullptr, tot, B.stat_sum, B.stat_sq,
                                   bn_mid_g + (size_t)i * EMB, bn_mid_b + (size_t)i * EMB,
                                   N_NODES, EMB, 1);
  }
}

extern "C" void kernel_launch(void* const* d_in, const int* in_sizes, int n_in,
                              void* d_out, int out_size, void* d_ws, size_t ws_size,
                              hipStream_t stream) {
  (void)n_in; (void)out_size; (void)ws_size;
  const float* x        = (const float*)d_in[0];
  const int*   ei       = (const int*)d_in[1];
  const int*   batch    = (const int*)d_in[2];
  const float* x_neg    = (const float*)d_in[3];
  const int*   ei_neg   = (const int*)d_in[4];
  const float* W_in_rel = (const float*)d_in[6];
  const float* W_in_root= (const float*)d_in[8];
  const float* W_mid_rel= (const float*)d_in[11];
  const float* W_mid_root=(const float*)d_in[13];
  const float* W_out    = (const float*)d_in[16];
  const float* b_out    = (const float*)d_in[17];
  const float* bn_out_g = (const float*)d_in[18];
  const float* bn_out_b = (const float*)d_in[19];
  const float* W_disc   = (const float*)d_in[20];

  int nE = in_sizes[1] / 2;
  const int* src  = ei;
  const int* dst  = ei + nE;
  const int* srcN = ei_neg;
  const int* dstN = ei_neg + nE;

  const size_t PANEL = (size_t)N_NODES * EMB;     // 16,777,216
  Bufs B;
  {
    float* w = (float*)d_ws;
    size_t off = 0;
    B.agg     = w + off; off += PANEL;
    B.lin     = w + off; off += PANEL;
    B.totP    = w + off; off += PANEL;
    B.totN    = w + off; off += PANEL;
    B.sums    = w + off; off += (size_t)N_GRAPH * EMB;
    B.summary = w + off; off += (size_t)N_GRAPH * EMB;
    B.wsb     = w + off; off += (size_t)EMB * EMB;
    B.outlin  = w + off; off += (size_t)N_GRAPH * TGT;
    B.stat_sum= w + off; off += 256;
    B.stat_sq = w + off; off += 256;
    B.cnts    = w + off; off += 256;
    B.lossA   = w + off; off += 4;
    __bf16* bw = (__bf16*)(w + off);
    size_t boff = 0;
    B.x_bf      = bw + boff; boff += (size_t)N_NODES * F_IN;
    B.xneg_bf   = bw + boff; boff += (size_t)N_NODES * F_IN;
    B.h_bf      = bw + boff; boff += PANEL;
    B.tot_bf    = bw + boff; boff += PANEL;
    B.summary_bf= bw + boff; boff += (size_t)N_GRAPH * EMB;
    B.wsb_bf    = bw + boff; boff += (size_t)EMB * EMB;
    B.Wir_bf    = bw + boff; boff += (size_t)F_IN * EMB;
    B.Wirt_bf   = bw + boff; boff += (size_t)F_IN * EMB;
    B.Wmr_bf    = bw + boff; boff += (size_t)NLAYERS * EMB * EMB;
    B.Wmrt_bf   = bw + boff; boff += (size_t)NLAYERS * EMB * EMB;
    B.Wout_bf   = bw + boff; boff += (size_t)EMB * TGT;
    B.Wdisc_bf  = bw + boff; boff += (size_t)EMB * EMB;
  }

  // -------- one-shot conversions (inputs + weights -> bf16 at rest) --------
  launch_cvt(x,         B.x_bf,    (size_t)N_NODES * F_IN, stream);
  launch_cvt(x_neg,     B.xneg_bf, (size_t)N_NODES * F_IN, stream);
  launch_cvt(W_in_rel,  B.Wir_bf,  (size_t)F_IN * EMB, stream);
  launch_cvt(W_in_root, B.Wirt_bf, (size_t)F_IN * EMB, stream);
  launch_cvt(W_mid_rel, B.Wmr_bf,  (size_t)NLAYERS * EMB * EMB, stream);
  launch_cvt(W_mid_root,B.Wmrt_bf, (size_t)NLAYERS * EMB * EMB, stream);
  launch_cvt(W_out,     B.Wout_bf, (size_t)EMB * TGT, stream);
  launch_cvt(W_disc,    B.Wdisc_bf,(size_t)EMB * EMB, stream);

  // -------- encoders --------
  run_encoder(B.x_bf,    x,     src,  dst,  nE, d_in, B, B.totP, stream);
  run_encoder(B.xneg_bf, x_neg, srcN, dstN, nE, d_in, B, B.totN, stream);

  // -------- global mean pool (pos) --------
  launch_zero(B.sums, (size_t)N_GRAPH * EMB, stream);
  launch_zero(B.cnts, 256, stream);
  k_pool_accum<<<8192, 256, 0, stream>>>(B.totP, batch, B.sums, N_NODES, EMB);
  k_pool_counts<<<(N_NODES + 255) / 256, 256, 0, stream>>>(batch, B.cnts, N_NODES);
  k_pool_div<<<(N_GRAPH * EMB + 255) / 256, 256, 0, stream>>>(B.sums, B.cnts, B.summary,
                                                              B.summary_bf, N_GRAPH, EMB);

  // -------- dense out layer: relu(bn(summary @ W_out + b_out)) -> d_out[0:32768] --------
  k_gemm_enc<<<gemm_blocks(N_GRAPH, TGT), 256, 0, stream>>>(
      B.summary, B.Wout_bf, EMB, nullptr, nullptr, 0, b_out, B.outlin, N_GRAPH, TGT);
  launch_zero(B.stat_sum, 512, stream);
  k_bn_stats<<<64, TGT, 0, stream>>>(B.outlin, N_GRAPH, TGT, B.stat_sum, B.stat_sq);
  k_bn_relu<<<128, 256, 0, stream>>>(B.outlin, nullptr, (float*)d_out, nullptr,
                                     B.stat_sum, B.stat_sq, bn_out_g, bn_out_b,
                                     N_GRAPH, TGT, 0);

  // -------- discriminator: ws = W_disc @ summary ; fused log-sigmoid loss --------
  k_gemm_bf<<<gemm_blocks(EMB, EMB), 256, 0, stream>>>(
      B.Wdisc_bf, B.summary_bf, EMB, B.wsb, EMB, EMB);
  launch_cvt(B.wsb, B.wsb_bf, (size_t)EMB * EMB, stream);
  launch_zero(B.lossA, 1, stream);
  {
    int blocks = gemm_blocks(N_NODES, EMB);
    launch_cvt(B.totP, B.tot_bf, PANEL, stream);
    k_loss_gemm<<<blocks, 256, 0, stream>>>(B.tot_bf, B.wsb_bf, N_NODES, EMB, EMB, B.lossA, 0);
    launch_cvt(B.totN, B.tot_bf, PANEL, stream);
    k_loss_gemm<<<blocks, 256, 0, stream>>>(B.tot_bf, B.wsb_bf, N_NODES, EMB, EMB, B.lossA, 1);
  }
  k_finalize_loss<<<1, 32, 0, stream>>>(B.lossA, (float*)d_out + (size_t)N_GRAPH * TGT,
                                        1.0f / ((float)N_NODES * (float)EMB));
}
